// SingleHeadAttention_63376537420281
// MI455X (gfx1250) — matrix operand order repaired and checked by reading, not compiled, not run on hardware
//
#include <hip/hip_runtime.h>
#include <hip/hip_bf16.h>
#include <math.h>

// ---------------------------------------------------------------------------
// Single-head causal attention with RoPE for MI455X (gfx1250, wave32, WMMA).
//   B=16, S=2048, D=128.  fp32 in/out, fp16 WMMA (f32 accumulate) internally.
//
// Roofline: ~20 GFLOP of matmul vs ~50 MB of unavoidable HBM traffic ->
// compute-bound; use v_wmma_f32_16x16x32_f16 throughout. K/V per batch is
// 0.5 MB fp16 -> lives in 192MB L2 across all q-tiles, so B-fragments are
// loaded straight from global (16B contiguous per lane), no K/V LDS staging.
// LDS is used only for the P (probabilities) C-layout -> A-layout reshuffle.
// ---------------------------------------------------------------------------

typedef __attribute__((ext_vector_type(16))) _Float16 v16h;
typedef __attribute__((ext_vector_type(8)))  _Float16 v8h;
typedef __attribute__((ext_vector_type(8)))  float    v8f;

constexpr int Bn = 16;
constexpr int Sn = 2048;
constexpr int Dn = 128;

// --- WMMA fragment loaders (wave32, 16x16x32 f16 shapes) -------------------
// A matrix 16x32 (MxK), row-major source with leading dim `ld` (f16 units).
// Lane L<16 : row = L,    K elements {0..7, 16..23}
// Lane L>=16: row = L-16, K elements {8..15, 24..31}
__device__ __forceinline__ v16h load_frag_a(const _Float16* base, int ld,
                                            int row_base, int k_base, int lane) {
  const int m    = lane & 15;
  const int half = lane >> 4;
  const _Float16* p = base + (size_t)(row_base + m) * ld + k_base + half * 8;
  v8h lo = *(const v8h*)(p);
  v8h hi = *(const v8h*)(p + 16);
  v16h a;
#pragma unroll
  for (int i = 0; i < 8; ++i) { a[i] = lo[i]; a[8 + i] = hi[i]; }
  return a;
}

// B matrix 32x16 (KxN). Source rows indexed by N, contiguous over K
// (i.e. src[n][k]) with leading dim `ld`.
// Lane L: n = L&15, K elements {0..7, 16..23} (L<16) or {8..15, 24..31} (L>=16).
__device__ __forceinline__ v16h load_frag_b(const _Float16* base, int ld,
                                            int n_base, int k_base, int lane) {
  const int n    = lane & 15;
  const int half = lane >> 4;
  const _Float16* p = base + (size_t)(n_base + n) * ld + k_base + half * 8;
  v8h lo = *(const v8h*)(p);
  v8h hi = *(const v8h*)(p + 16);
  v16h b;
#pragma unroll
  for (int i = 0; i < 8; ++i) { b[i] = lo[i]; b[8 + i] = hi[i]; }
  return b;
}

// --- Kernel 1: fp32 -> fp16 conversion of x and the three weight matrices --
__global__ __launch_bounds__(256) void convert_kernel(
    const float* __restrict__ x, const float* __restrict__ Wq,
    const float* __restrict__ Wk, const float* __restrict__ Wv,
    _Float16* __restrict__ xh, _Float16* __restrict__ Wh) {
  const int i  = blockIdx.x * blockDim.x + threadIdx.x;
  const int nx = Bn * Sn * Dn;
  const int nw = Dn * Dn;
  if (i < nx) xh[i] = (_Float16)x[i];
  if (i < nw) {
    Wh[i]          = (_Float16)Wq[i];
    Wh[nw + i]     = (_Float16)Wk[i];
    Wh[2 * nw + i] = (_Float16)Wv[i];
  }
}

// --- Kernel 2: QKV projection + bias + RoPE -------------------------------
// One wave computes one 16-row tile of one of {Q,K,V}. GEMM is x @ W^T, so
// the contraction dim (d) is contiguous in both x rows (A frags) and W rows
// (B frags). Q,K stored fp16 row-major [B*S, D]; V stored transposed
// [B, D, S] so attention-phase P@V B-fragments are contiguous.
__global__ __launch_bounds__(256) void qkv_rope_kernel(
    const _Float16* __restrict__ xh, const _Float16* __restrict__ Wh,
    const float* __restrict__ bq, const float* __restrict__ bk,
    const float* __restrict__ bv,
    _Float16* __restrict__ Qh, _Float16* __restrict__ Kh,
    _Float16* __restrict__ Vt) {
  const int lane = threadIdx.x & 31;
  const int wave = threadIdx.x >> 5;
  const int w    = blockIdx.x * 8 + wave;          // 0 .. 3*(B*S/16)-1
  const int mat  = w / (Bn * Sn / 16);             // 0=Q 1=K 2=V
  const int rt   = w % (Bn * Sn / 16);
  const int row_base = rt * 16;                    // row into [B*S, D]
  const int half = lane >> 4;
  const int n    = lane & 15;

  v16h af[4];
#pragma unroll
  for (int kc = 0; kc < 4; ++kc) af[kc] = load_frag_a(xh, Dn, row_base, kc * 32, lane);

  const _Float16* W = Wh + (size_t)mat * Dn * Dn;
  const float* bias = (mat == 0) ? bq : (mat == 1) ? bk : bv;
  const int b_idx = row_base / Sn;
  const int s0    = row_base % Sn;

#pragma unroll
  for (int ct = 0; ct < 8; ++ct) {
    v8f acc = {};
#pragma unroll
    for (int kc = 0; kc < 4; ++kc) {
      v16h bf = load_frag_b(W, Dn, ct * 16, kc * 32, lane);
      acc = __builtin_amdgcn_wmma_f32_16x16x32_f16(false, af[kc], false, bf,
                                                   (short)0, acc, false, false);
    }
    const int   col = ct * 16 + n;
    const float bb  = bias[col];
#pragma unroll
    for (int j = 0; j < 8; ++j) acc[j] += bb;

    if (mat < 2) {
      // RoPE: column pairs (2i, 2i+1) sit in adjacent lanes of the C tile.
      const float inv_freq =
          __powf(10000.0f, -(float)(2 * (col >> 1)) * (1.0f / 128.0f));
#pragma unroll
      for (int j = 0; j < 8; ++j) {
        float own   = acc[j];
        float other = __shfl_xor(own, 1, 32);
        float srow  = (float)(s0 + j + 8 * half);
        float th    = srow * inv_freq;
        float c = __cosf(th), sn = __sinf(th);
        acc[j] = ((col & 1) == 0) ? (own * c + other * sn)
                                  : (-other * sn + own * c);
      }
      _Float16* dst = (mat == 0) ? Qh : Kh;
#pragma unroll
      for (int j = 0; j < 8; ++j)
        dst[(size_t)(row_base + j + 8 * half) * Dn + col] = (_Float16)acc[j];
    } else {
      // V transposed: lane's 8 rows are contiguous along S -> one b128 store.
      v8h pk;
#pragma unroll
      for (int j = 0; j < 8; ++j) pk[j] = (_Float16)acc[j];
      *(v8h*)(Vt + ((size_t)b_idx * Dn + col) * Sn + s0 + 8 * half) = pk;
    }
  }
}

// --- Kernel 3: causal flash attention -------------------------------------
// Block = 8 waves; wave w owns 16 query rows. 32-key tiles, online softmax.
__global__ __launch_bounds__(256) void flash_attn_kernel(
    const _Float16* __restrict__ Qh, const _Float16* __restrict__ Kh,
    const _Float16* __restrict__ Vt, float* __restrict__ out) {
  __shared__ alignas(16) _Float16 pbuf[8][16 * 32];  // wave-private P tiles

  const int lane = threadIdx.x & 31;
  const int wave = threadIdx.x >> 5;
  const int b    = blockIdx.x >> 4;                 // S/128 = 16 q-tiles
  const int qt   = blockIdx.x & 15;
  const int qr   = qt * 128 + wave * 16;            // first query row of wave
  const int half = lane >> 4;
  const int n    = lane & 15;

  const _Float16* Qb = Qh + (size_t)b * Sn * Dn;
  const _Float16* Kb = Kh + (size_t)b * Sn * Dn;
  const _Float16* Vb = Vt + (size_t)b * Dn * Sn;

  v16h qf[4];
#pragma unroll
  for (int kc = 0; kc < 4; ++kc) qf[kc] = load_frag_a(Qb, Dn, qr, kc * 32, lane);

  float mrow[8], lrow[8];
  v8f   oacc[8];
#pragma unroll
  for (int j = 0; j < 8; ++j) { mrow[j] = -1e30f; lrow[j] = 0.0f; }
#pragma unroll
  for (int t = 0; t < 8; ++t) oacc[t] = (v8f){};

  const float scale  = 0.08838834764831845f;        // 1/sqrt(128)
  const int   kt_max = (qr + 15) >> 5;              // causal tile bound
  _Float16*   pw     = &pbuf[wave][0];

  for (int kt = 0; kt <= kt_max; ++kt) {
    const int k_base = kt * 32;

    // --- scores: Q(16x128) @ K^T -> 16x32, 8 WMMAs ---
    v8f sc[2];
#pragma unroll
    for (int ns = 0; ns < 2; ++ns) {
      v8f acc = {};
#pragma unroll
      for (int kc = 0; kc < 4; ++kc) {
        v16h bf = load_frag_b(Kb, Dn, k_base + ns * 16, kc * 32, lane);
        acc = __builtin_amdgcn_wmma_f32_16x16x32_f16(false, qf[kc], false, bf,
                                                     (short)0, acc, false, false);
      }
      const bool edge = (k_base + 31 > qr);
#pragma unroll
      for (int j = 0; j < 8; ++j) {
        float s = acc[j] * scale;
        if (edge) {
          int colg = k_base + ns * 16 + n;
          int rowg = qr + j + 8 * half;
          if (colg > rowg) s = -1e30f;
        }
        acc[j] = s;
      }
      sc[ns] = acc;
    }

    // --- online softmax: row max across 16 lanes of each half ---
    float mnew[8], alpha[8];
#pragma unroll
    for (int j = 0; j < 8; ++j) {
      float mx = fmaxf(sc[0][j], sc[1][j]);
#pragma unroll
      for (int msk = 1; msk < 16; msk <<= 1)
        mx = fmaxf(mx, __shfl_xor(mx, msk, 32));
      mnew[j]  = fmaxf(mrow[j], mx);
      alpha[j] = __expf(mrow[j] - mnew[j]);
      mrow[j]  = mnew[j];
    }

    // --- P = exp(s - m); write fp16 P to LDS in row-major 16x32 ---
    float psum[8];
#pragma unroll
    for (int j = 0; j < 8; ++j) psum[j] = 0.0f;
#pragma unroll
    for (int ns = 0; ns < 2; ++ns) {
#pragma unroll
      for (int j = 0; j < 8; ++j) {
        float p = __expf(sc[ns][j] - mnew[j]);
        psum[j] += p;
        pw[(j + 8 * half) * 32 + ns * 16 + n] = (_Float16)p;
      }
    }
#pragma unroll
    for (int j = 0; j < 8; ++j) {
      float s = psum[j];
#pragma unroll
      for (int msk = 1; msk < 16; msk <<= 1) s += __shfl_xor(s, msk, 32);
      lrow[j] = lrow[j] * alpha[j] + s;
    }
#pragma unroll
    for (int t = 0; t < 8; ++t)
#pragma unroll
      for (int j = 0; j < 8; ++j) oacc[t][j] *= alpha[j];

    // --- O += P(16x32) @ V(32x128): 8 WMMAs; P reloaded in A layout ---
    v16h pa = load_frag_a(pw, 32, 0, 0, lane);
#pragma unroll
    for (int dt = 0; dt < 8; ++dt) {
      v16h bf = load_frag_b(Vb, Sn, dt * 16, k_base, lane);
      oacc[dt] = __builtin_amdgcn_wmma_f32_16x16x32_f16(false, pa, false, bf,
                                                        (short)0, oacc[dt],
                                                        false, false);
    }
  }

  // --- finalize: O /= l, fp32 store ---
  float inv_l[8];
#pragma unroll
  for (int j = 0; j < 8; ++j) inv_l[j] = 1.0f / lrow[j];
  float* ob = out + ((size_t)b * Sn + qr) * Dn;
#pragma unroll
  for (int dt = 0; dt < 8; ++dt)
#pragma unroll
    for (int j = 0; j < 8; ++j)
      ob[(size_t)(j + 8 * half) * Dn + dt * 16 + n] = oacc[dt][j] * inv_l[j];
}

// ---------------------------------------------------------------------------
extern "C" void kernel_launch(void* const* d_in, const int* in_sizes, int n_in,
                              void* d_out, int out_size, void* d_ws,
                              size_t ws_size, hipStream_t stream) {
  (void)in_sizes; (void)n_in; (void)out_size; (void)ws_size;
  const float* x  = (const float*)d_in[0];
  const float* Wq = (const float*)d_in[1];
  const float* bq = (const float*)d_in[2];
  const float* Wk = (const float*)d_in[3];
  const float* bk = (const float*)d_in[4];
  const float* Wv = (const float*)d_in[5];
  const float* bv = (const float*)d_in[6];
  float* out = (float*)d_out;

  // Workspace layout (fp16): xh | Qh | Kh | Vt | Wh  (~33.7 MB total)
  char* ws = (char*)d_ws;
  const size_t nBSD = (size_t)Bn * Sn * Dn;        // 4,194,304 elements
  _Float16* xh = (_Float16*)(ws);
  _Float16* Qh = (_Float16*)(ws + 2 * nBSD);
  _Float16* Kh = (_Float16*)(ws + 4 * nBSD);
  _Float16* Vt = (_Float16*)(ws + 6 * nBSD);
  _Float16* Wh = (_Float16*)(ws + 8 * nBSD);

  convert_kernel<<<(int)((nBSD + 255) / 256), 256, 0, stream>>>(x, Wq, Wk, Wv,
                                                                xh, Wh);
  qkv_rope_kernel<<<(3 * (Bn * Sn / 16)) / 8, 256, 0, stream>>>(
      xh, Wh, bq, bk, bv, Qh, Kh, Vt);
  flash_attn_kernel<<<Bn * (Sn / 128), 256, 0, stream>>>(Qh, Kh, Vt, out);
}
